// EPTLayer_42116449305116
// MI455X (gfx1250) — matrix operand order repaired and checked by reading, not compiled, not run on hardware
//
#include <hip/hip_runtime.h>
#include <cstdint>
#include <cstddef>

// ---------------------------------------------------------------------------
// EPT layer for MI455X (gfx1250): bf16 WMMA GEMMs + fused flash attention
// with double-buffered TDM (tensor_load_to_lds) K/V tile staging.
// B=2, N=1024, D_HID=512, N_HEADS=16, D_HEAD=32, D_FFN=2048.
// ---------------------------------------------------------------------------

typedef __bf16 bf16;
typedef __attribute__((ext_vector_type(16))) __bf16 v16bf;
typedef __attribute__((ext_vector_type(8)))  __bf16 v8bf;
typedef __attribute__((ext_vector_type(8)))  float  v8f;
typedef unsigned int v4u __attribute__((ext_vector_type(4)));
typedef int          v4i __attribute__((ext_vector_type(4)));
typedef int          v8i __attribute__((ext_vector_type(8)));

#define BB    2
#define NSEQ  1024
#define DHID  512
#define NH    16
#define DHEAD 32
#define DFF   2048
#define BN    (BB * NSEQ)        // 2048
#define BN3   (BN * 3)           // 6144
#define QKW   (NH * 4 * DHEAD)   // 2048 = per-token q/k width, also V_attn width
#define ATTN_FACTOR 0.08838834764831845f  // 0.5 / sqrt(32)

#if __has_builtin(__builtin_amdgcn_tensor_load_to_lds) && \
    __has_builtin(__builtin_amdgcn_s_wait_tensorcnt)
#define EPT_USE_TDM 1
#else
#define EPT_USE_TDM 0
#endif

// ----------------------------- WMMA helpers --------------------------------

__device__ __forceinline__ v8f wmma_bf16(v16bf a, v16bf b, v8f c) {
  // D = A(16x32) * B(32x16) + C, f32 accumulate
  return __builtin_amdgcn_wmma_f32_16x16x32_bf16(false, a, false, b, (short)0, c,
                                                 false, false);
}

// A fragment (16x32 bf16, row-major source, ld in elements).
// Lanes 0-15 hold row M=lane, K=0..15; lanes 16-31 hold
// row M=lane-16, K=16..31 (contiguous, matching load_frag_b).
__device__ __forceinline__ v16bf load_frag_a(const bf16* __restrict__ base, int ld) {
  const int lane = threadIdx.x & 31;
  const int row  = lane & 15;
  const int kh   = lane >> 4;
  const bf16* p = base + (size_t)row * ld + kh * 16;
  v8bf lo = *(const v8bf*)(p);
  v8bf hi = *(const v8bf*)(p + 8);
  v16bf a;
#pragma unroll
  for (int i = 0; i < 8; ++i) { a[i] = lo[i]; a[i + 8] = hi[i]; }
  return a;
}

// B fragment (32x16 bf16). Source S row-major with S[n][k] = B[k][n].
// Lane n holds column n&15; half-wave selects K=0..15 vs 16..31, contiguous.
__device__ __forceinline__ v16bf load_frag_b(const bf16* __restrict__ base, int ld) {
  const int lane = threadIdx.x & 31;
  const int col  = lane & 15;
  const int kh   = lane >> 4;
  return *(const v16bf*)(base + (size_t)col * ld + kh * 16);
}

// B fragment where B[k][n] = src[k*rstride + n] (K strided across rows).
__device__ __forceinline__ v16bf load_frag_b_strided(const bf16* __restrict__ base,
                                                     int rstride) {
  const int lane = threadIdx.x & 31;
  const int col  = lane & 15;
  const int kh   = lane >> 4;
  const bf16* p = base + col + (size_t)(kh * 16) * rstride;
  v16bf b;
#pragma unroll
  for (int t = 0; t < 16; ++t) b[t] = p[(size_t)t * rstride];
  return b;
}

// ------------------------- cross-lane reductions ---------------------------

__device__ __forceinline__ float reduce_sum32(float v) {
#pragma unroll
  for (int off = 16; off >= 1; off >>= 1) v += __shfl_xor(v, off, 32);
  return v;
}
// Reduce across the 16 lanes of a half-wave (xor masks 1..8 stay in-half).
__device__ __forceinline__ float half_reduce_max16(float v) {
#pragma unroll
  for (int off = 8; off >= 1; off >>= 1) v = fmaxf(v, __shfl_xor(v, off, 32));
  return v;
}
__device__ __forceinline__ float half_reduce_sum16(float v) {
#pragma unroll
  for (int off = 8; off >= 1; off >>= 1) v += __shfl_xor(v, off, 32);
  return v;
}

// ------------------------------ TDM staging --------------------------------

#if EPT_USE_TDM
// DMA a 32-row x 128-elem bf16 tile (row stride `row_stride_elems`) from
// global memory into LDS at byte address lds_byte_addr. D# per ISA ch.8:
// group0 = {count, lds_addr, global_addr, type=2}; group1 = {data_size=2B,
// tensor dims (large, no OOB), tile_dim0=128, tile_dim1=32, dim0 stride}.
__device__ __forceinline__ void tdm_load_tile_32x128(const bf16* gsrc,
                                                     unsigned lds_byte_addr,
                                                     int row_stride_elems) {
  const unsigned long long ga = (unsigned long long)(size_t)gsrc;
  v4u g0;
  g0.x = 1u;                                               // count=1, user mode
  g0.y = lds_byte_addr;                                    // lds_addr
  g0.z = (unsigned)(ga & 0xffffffffull);                   // global_addr[31:0]
  g0.w = (unsigned)((ga >> 32) & 0x01ffffffull) | (2u << 30);  // [56:32]|type=2
  v8i g1;
  g1[0] = 1 << 16;                 // workgroup_mask=0, data_size=1 (2 bytes)
  g1[1] = 0;                       // no atomic barrier; tensor_dim0[15:0]=0
  g1[2] = 0x10;                    // tensor_dim0=1<<20 ; tensor_dim1[15:0]=0
  g1[3] = 0x10 | (128 << 16);      // tensor_dim1=1<<20 ; tile_dim0=128
  g1[4] = 32;                      // tile_dim1=32 ; tile_dim2=0
  g1[5] = row_stride_elems;        // tensor_dim0_stride[31:0]
  g1[6] = 0;                       // stride hi, tensor_dim1_stride lo
  g1[7] = 0;
  const v4i z4 = {0, 0, 0, 0};
#if __clang_major__ >= 23
  const v8i z8 = {0, 0, 0, 0, 0, 0, 0, 0};
  __builtin_amdgcn_tensor_load_to_lds(g0, g1, z4, z4, z8, 0);
#else
  __builtin_amdgcn_tensor_load_to_lds(g0, g1, z4, z4, 0);
#endif
}
#endif

// ------------------------------- kernels -----------------------------------

__global__ void f32_to_bf16_kernel(const float* __restrict__ x,
                                   bf16* __restrict__ y, size_t n) {
  for (size_t i = (size_t)blockIdx.x * blockDim.x + threadIdx.x; i < n;
       i += (size_t)gridDim.x * blockDim.x)
    y[i] = (bf16)x[i];
}

// LayerNorm: one wave per row of length DHID.
__global__ __launch_bounds__(32)
void layernorm_kernel(const float* __restrict__ X, const float* __restrict__ g,
                      const float* __restrict__ bta, bf16* __restrict__ Yb,
                      float* __restrict__ Yf) {
  const int row  = blockIdx.x;
  const int lane = threadIdx.x;
  const float* x = X + (size_t)row * DHID;
  float s = 0.f;
  for (int i = lane; i < DHID; i += 32) s += x[i];
  const float mean = reduce_sum32(s) * (1.0f / DHID);
  float v = 0.f;
  for (int i = lane; i < DHID; i += 32) { float d = x[i] - mean; v += d * d; }
  const float rstd = rsqrtf(reduce_sum32(v) * (1.0f / DHID) + 1e-5f);
  for (int i = lane; i < DHID; i += 32) {
    float y = (x[i] - mean) * rstd * g[i] + bta[i];
    if (Yb) Yb[(size_t)row * DHID + i] = (bf16)y;
    if (Yf) Yf[(size_t)row * DHID + i] = y;
  }
}

// Generic GEMM: C[M,N] = act(A[M,K] @ W[N,K]^T + bias) + resid
// 256 threads = 8 waves; wave tile 32x64 (2 A-frags x 4 B-frags, 8 wmma/K-step).
// grid = (N/64, M/256). act: 0 = none, 1 = SiLU.
__global__ __launch_bounds__(256)
void gemm_bf16_kernel(const bf16* __restrict__ A, const bf16* __restrict__ W,
                      const float* __restrict__ bias,
                      const float* __restrict__ resid,
                      float* __restrict__ Cf, bf16* __restrict__ Cb,
                      int M, int N, int K, int act) {
  const int wave = threadIdx.x >> 5;
  const int m0 = blockIdx.y * 256 + wave * 32;
  const int n0 = blockIdx.x * 64;
  const int lane = threadIdx.x & 31;
  const int col = lane & 15;
  const int rb  = (lane >> 4) * 8;

  v8f acc[2][4] = {};
  for (int k0 = 0; k0 < K; k0 += 32) {
    if (k0 + 32 < K) {  // emits global_prefetch_b8
      __builtin_prefetch(A + (size_t)m0 * K + k0 + 32, 0, 1);
      __builtin_prefetch(W + (size_t)n0 * K + k0 + 32, 0, 1);
    }
    v16bf a0 = load_frag_a(A + (size_t)m0 * K + k0, K);
    v16bf a1 = load_frag_a(A + (size_t)(m0 + 16) * K + k0, K);
#pragma unroll
    for (int j = 0; j < 4; ++j) {
      v16bf b = load_frag_b(W + (size_t)(n0 + j * 16) * K + k0, K);
      acc[0][j] = wmma_bf16(a0, b, acc[0][j]);
      acc[1][j] = wmma_bf16(a1, b, acc[1][j]);
    }
  }
#pragma unroll
  for (int i = 0; i < 2; ++i) {
#pragma unroll
    for (int j = 0; j < 4; ++j) {
      const int n = n0 + j * 16 + col;
      const float bs = bias ? bias[n] : 0.f;
#pragma unroll
      for (int r = 0; r < 8; ++r) {
        const int m = m0 + i * 16 + rb + r;
        float v = acc[i][j][r] + bs;
        if (act == 1) v = v / (1.f + __expf(-v));  // SiLU
        if (resid) v += resid[(size_t)m * N + n];
        if (Cf) Cf[(size_t)m * N + n] = v;
        if (Cb) Cb[(size_t)m * N + n] = (bf16)v;
      }
    }
  }
}

// V_attn[bn, h, 0:32] = hv[bn, h*32+j]; V_attn[bn, h, 32+c*32+j] = Vv[(bn*3+c), h*32+j]
__global__ void assemble_vattn_kernel(const float* __restrict__ hv,
                                      const float* __restrict__ vv,
                                      bf16* __restrict__ out) {
  const size_t total = (size_t)BN * NH * 128;
  for (size_t i = (size_t)blockIdx.x * blockDim.x + threadIdx.x; i < total;
       i += (size_t)gridDim.x * blockDim.x) {
    const int bn = (int)(i >> 11);
    const int rem = (int)(i & 2047);
    const int h = rem >> 7, s = rem & 127;
    float val;
    if (s < 32) val = hv[(size_t)bn * DHID + h * 32 + s];
    else {
      const int c = (s - 32) >> 5, j = s & 31;
      val = vv[((size_t)bn * 3 + c) * DHID + h * 32 + j];
    }
    out[i] = (bf16)val;
  }
}

// Fused flash attention. 4 waves per block; block owns a 64-row tile of one
// (b, h); each wave computes 16 rows. K/V tiles (32x128 bf16 = 8KB each) are
// double-buffered in LDS and staged by the Tensor Data Mover: tile i+1 is
// DMA'd while tile i is computed; s_wait_tensorcnt(2) lets the next tile's
// pair stay in flight (TDM ops of one wave complete in order). Fallback is a
// coalesced manual copy. grid = B*NH*(N/64).
__global__ __launch_bounds__(128)
void attention_kernel(const bf16* __restrict__ Q, const bf16* __restrict__ Kb,
                      const bf16* __restrict__ Vb, const float* __restrict__ rbf,
                      const float* __restrict__ Dm, const int* __restrict__ mask,
                      float* __restrict__ Out) {
  const int wg = blockIdx.x;
  const int nt = wg & 15;          // 64-row tile index (NSEQ/64 = 16)
  const int h  = (wg >> 4) & 15;
  const int b  = wg >> 8;
  const int wave = threadIdx.x >> 5;
  const int n0 = nt * 64 + wave * 16;
  const int lane = threadIdx.x & 31;
  const int col = lane & 15;
  const int rb  = (lane >> 4) * 8;

  __shared__ __attribute__((aligned(32))) bf16 kTile[2][32 * 128];
  __shared__ __attribute__((aligned(32))) bf16 vTile[2][32 * 128];
  __shared__ __attribute__((aligned(32))) bf16 pLds[4][16 * 32];
  bf16* pw = pLds[wave];

  // Q fragments: rows n0..n0+15, head dims 0..127 (4 x K=32)
  const bf16* qbase = Q + ((size_t)(b * NSEQ + n0)) * QKW + h * 128;
  v16bf qa[4];
#pragma unroll
  for (int kb = 0; kb < 4; ++kb) qa[kb] = load_frag_a(qbase + kb * 32, QKW);

  v8f o[8] = {};
  float mrun[8], lrun[8];
#pragma unroll
  for (int r = 0; r < 8; ++r) { mrun[r] = -1e30f; lrun[r] = 0.f; }

  const float* rbf_bh = rbf + ((size_t)(b * NH + h)) * NSEQ * NSEQ;
  const float* D_b    = Dm + (size_t)b * NSEQ * NSEQ;
  const int*   msk    = mask + (size_t)b * NSEQ;
  const bf16*  kRow0  = Kb + (size_t)b * NSEQ * QKW + h * 128;
  const bf16*  vRow0  = Vb + (size_t)b * NSEQ * QKW + h * 128;

#if EPT_USE_TDM
  if (wave == 0) {  // prologue DMA: tile 0 into buffer 0 (EXEC ignored by TDM)
    tdm_load_tile_32x128(kRow0, (unsigned)(size_t)kTile[0], QKW);
    tdm_load_tile_32x128(vRow0, (unsigned)(size_t)vTile[0], QKW);
  }
#endif

  for (int it = 0; it < NSEQ / 32; ++it) {
    const int m0 = it * 32;
    const int cur = it & 1;
#if EPT_USE_TDM
    if (wave == 0) {
      if (it + 1 < NSEQ / 32) {  // kick DMA for next tile, then wait current
        const int nxt = (it + 1) & 1;
        tdm_load_tile_32x128(kRow0 + (size_t)(m0 + 32) * QKW,
                             (unsigned)(size_t)kTile[nxt], QKW);
        tdm_load_tile_32x128(vRow0 + (size_t)(m0 + 32) * QKW,
                             (unsigned)(size_t)vTile[nxt], QKW);
        __builtin_amdgcn_s_wait_tensorcnt(2);  // next pair may stay in flight
      } else {
        __builtin_amdgcn_s_wait_tensorcnt(0);
      }
    }
    __syncthreads();
#else
    // coalesced manual staging: 512 chunks of 16B per tile, 4 per thread
    for (int c = threadIdx.x; c < 512; c += 128) {
      const int row = c >> 4, o8 = (c & 15) * 8;
      *(v8bf*)(kTile[cur] + row * 128 + o8) =
          *(const v8bf*)(kRow0 + (size_t)(m0 + row) * QKW + o8);
      *(v8bf*)(vTile[cur] + row * 128 + o8) =
          *(const v8bf*)(vRow0 + (size_t)(m0 + row) * QKW + o8);
    }
    __syncthreads();
#endif
    const bf16* kT = kTile[cur];
    const bf16* vT = vTile[cur];

    // S = q @ k^T for 32 keys (two 16x16 tiles, K=128 -> 4 steps each)
    v8f s0 = {}, s1 = {};
#pragma unroll
    for (int kb = 0; kb < 4; ++kb) {
      v16bf b0 = load_frag_b(kT + kb * 32, 128);
      v16bf b1 = load_frag_b(kT + 16 * 128 + kb * 32, 128);
      s0 = wmma_bf16(qa[kb], b0, s0);
      s1 = wmma_bf16(qa[kb], b1, s1);
    }
    // bias + mask + online softmax (rows live in half-waves; columns = lanes)
#pragma unroll
    for (int r = 0; r < 8; ++r) {
      const int row = n0 + rb + r;
      const int c0 = m0 + col, c1 = m0 + 16 + col;
      const size_t ro = (size_t)row * NSEQ;
      float b0v = rbf_bh[ro + c0] + D_b[ro + c0];
      float b1v = rbf_bh[ro + c1] + D_b[ro + c1];
      if (msk[c0] == 0) b0v = -1e30f;
      if (msk[c1] == 0) b1v = -1e30f;
      const float p0 = s0[r] * ATTN_FACTOR + b0v;
      const float p1 = s1[r] * ATTN_FACTOR + b1v;
      const float tmax = half_reduce_max16(fmaxf(p0, p1));
      const float nm = fmaxf(mrun[r], tmax);
      const float scale = __expf(mrun[r] - nm);
      const float e0 = __expf(p0 - nm);
      const float e1 = __expf(p1 - nm);
      const float rs = half_reduce_sum16(e0 + e1);
      lrun[r] = lrun[r] * scale + rs;
      mrun[r] = nm;
#pragma unroll
      for (int j = 0; j < 8; ++j) o[j][r] *= scale;
      pw[(rb + r) * 32 + col]      = (bf16)e0;
      pw[(rb + r) * 32 + 16 + col] = (bf16)e1;
    }
    // O += P(16x32) @ V_attn(32x128); per-wave P tile, shared V tile (LDS)
    v16bf pa = load_frag_a(pw, 32);
#pragma unroll
    for (int j = 0; j < 8; ++j) {
      v16bf vb = load_frag_b_strided(vT + j * 16, 128);
      o[j] = wmma_bf16(pa, vb, o[j]);
    }
    __syncthreads();  // all waves done with this buffer before it is re-DMA'd
  }
  // normalize and write res (bn, h, 128) f32
#pragma unroll
  for (int j = 0; j < 8; ++j) {
#pragma unroll
    for (int r = 0; r < 8; ++r) {
      const int row = n0 + rb + r;
      const float val = o[j][r] / lrun[r];
      Out[(((size_t)(b * NSEQ + row)) * NH + h) * 128 + j * 16 + col] = val;
    }
  }
}

// H_res[bn, h*32+j] = res[bn,h,j]; V_res[(bn*3+c), h*32+j] = res[bn,h,32+c*32+j]
__global__ void split_res_kernel(const float* __restrict__ res,
                                 bf16* __restrict__ hres, bf16* __restrict__ vres) {
  const size_t total = (size_t)BN * NH * 128;
  for (size_t i = (size_t)blockIdx.x * blockDim.x + threadIdx.x; i < total;
       i += (size_t)gridDim.x * blockDim.x) {
    const int bn = (int)(i >> 11);
    const int rem = (int)(i & 2047);
    const int h = rem >> 7, s = rem & 127;
    const float v = res[i];
    if (s < 32) hres[(size_t)bn * DHID + h * 32 + s] = (bf16)v;
    else {
      const int c = (s - 32) >> 5, j = s & 31;
      vres[((size_t)bn * 3 + c) * DHID + h * 32 + j] = (bf16)v;
    }
  }
}

// scaler[bn, 0:512] = Hn2; scaler[bn, 512+j] = || Vproj[bn, :, j] || over c
__global__ void scaler_kernel(const float* __restrict__ hn2,
                              const float* __restrict__ vproj,
                              bf16* __restrict__ out) {
  const size_t total = (size_t)BN * 1024;
  for (size_t i = (size_t)blockIdx.x * blockDim.x + threadIdx.x; i < total;
       i += (size_t)gridDim.x * blockDim.x) {
    const int bn = (int)(i >> 10);
    const int j = (int)(i & 1023);
    float val;
    if (j < DHID) val = hn2[(size_t)bn * DHID + j];
    else {
      const int jj = j - DHID;
      const size_t base = (size_t)bn * 3 * 1024 + jj;  // V1 = Vproj[..., :512]
      const float x0 = vproj[base], x1 = vproj[base + 1024], x2 = vproj[base + 2048];
      val = sqrtf(x0 * x0 + x1 * x1 + x2 * x2);
    }
    out[i] = (bf16)val;
  }
}

// H_out = s_out[:, :512] + H1 ; V_out = s_out[:, 512+j] * V2 + Vnew
__global__ void final_kernel(const float* __restrict__ sout,
                             const float* __restrict__ h1,
                             const float* __restrict__ vproj,
                             const float* __restrict__ vnew,
                             float* __restrict__ out) {
  const size_t hTotal = (size_t)BN * DHID;             // 1048576
  const size_t total = hTotal + (size_t)BN3 * DHID;    // + 3145728
  for (size_t i = (size_t)blockIdx.x * blockDim.x + threadIdx.x; i < total;
       i += (size_t)gridDim.x * blockDim.x) {
    if (i < hTotal) {
      const int bn = (int)(i >> 9);
      const int j = (int)(i & 511);
      out[i] = sout[(size_t)bn * 1024 + j] + h1[i];
    } else {
      const size_t k = i - hTotal;
      const int row = (int)(k >> 9);  // 0..6143 = bn*3+c
      const int j = (int)(k & 511);
      const int bn = row / 3;
      const float upd = sout[(size_t)bn * 1024 + DHID + j];
      const float v2 = vproj[(size_t)row * 1024 + DHID + j];
      out[i] = upd * v2 + vnew[k];
    }
  }
}

// ------------------------------- host side ---------------------------------

extern "C" void kernel_launch(void* const* d_in, const int* in_sizes, int n_in,
                              void* d_out, int out_size, void* d_ws, size_t ws_size,
                              hipStream_t stream) {
  const float* H    = (const float*)d_in[0];
  const float* V    = (const float*)d_in[1];
  const float* Dm   = (const float*)d_in[2];
  const float* rbf  = (const float*)d_in[3];
  const int*   msk  = (const int*)d_in[4];
  const float* Wq   = (const float*)d_in[5];
  const float* bq   = (const float*)d_in[6];
  const float* Wk   = (const float*)d_in[7];
  const float* bk   = (const float*)d_in[8];
  const float* Wv   = (const float*)d_in[9];
  const float* bv   = (const float*)d_in[10];
  const float* Wvv  = (const float*)d_in[11];
  const float* Wo   = (const float*)d_in[12];
  const float* bo   = (const float*)d_in[13];
  const float* Wvo  = (const float*)d_in[14];
  const float* ln1g = (const float*)d_in[15];
  const float* ln1b = (const float*)d_in[16];
  const float* Wlv  = (const float*)d_in[17];
  const float* W1   = (const float*)d_in[18];
  const float* b1   = (const float*)d_in[19];
  const float* W2   = (const float*)d_in[20];
  const float* b2   = (const float*)d_in[21];
  const float* ln2g = (const float*)d_in[22];
  const float* ln2b = (const float*)d_in[23];
  float* out = (float*)d_out;

  // workspace carve-out (256B aligned)
  char* p = (char*)d_ws;
  auto take = [&](size_t bytes) -> void* {
    void* r = (void*)p;
    p += (bytes + 255) & ~(size_t)255;
    return r;
  };
  bf16* wq_b   = (bf16*)take((size_t)QKW * DHID * 2);
  bf16* wk_b   = (bf16*)take((size_t)QKW * DHID * 2);
  bf16* wv_b   = (bf16*)take((size_t)DHID * DHID * 2);
  bf16* wvv_b  = (bf16*)take((size_t)DHID * DHID * 2);
  bf16* wo_b   = (bf16*)take((size_t)DHID * DHID * 2);
  bf16* wvo_b  = (bf16*)take((size_t)DHID * DHID * 2);
  bf16* wlv_b  = (bf16*)take((size_t)1024 * DHID * 2);
  bf16* w1_b   = (bf16*)take((size_t)DFF * 1024 * 2);
  bf16* w2_b   = (bf16*)take((size_t)1024 * DFF * 2);
  bf16* hn_b   = (bf16*)take((size_t)BN * DHID * 2);
  bf16* vin_b  = (bf16*)take((size_t)BN3 * DHID * 2);
  bf16* q_b    = (bf16*)take((size_t)BN * QKW * 2);
  bf16* k_b    = (bf16*)take((size_t)BN * QKW * 2);
  float* hv_f  = (float*)take((size_t)BN * DHID * 4);
  float* vv_f  = (float*)take((size_t)BN3 * DHID * 4);
  bf16* vatt_b = (bf16*)take((size_t)BN * QKW * 2);
  float* res_f = (float*)take((size_t)BN * QKW * 4);
  bf16* hres_b = (bf16*)take((size_t)BN * DHID * 2);
  bf16* vres_b = (bf16*)take((size_t)BN3 * DHID * 2);
  float* h1_f  = (float*)take((size_t)BN * DHID * 4);
  float* vn_f  = (float*)take((size_t)BN3 * DHID * 4);
  bf16* vn_b   = (bf16*)take((size_t)BN3 * DHID * 2);
  float* hn2_f = (float*)take((size_t)BN * DHID * 4);
  float* vpr_f = (float*)take((size_t)BN3 * 1024 * 4);
  bf16* scl_b  = (bf16*)take((size_t)BN * 1024 * 2);
  bf16* s1_b   = (bf16*)take((size_t)BN * DFF * 2);
  float* so_f  = (float*)take((size_t)BN * 1024 * 4);
  (void)ws_size; (void)n_in; (void)in_sizes; (void)out_size;

  const dim3 cvtG(2048), cvtB(256);
  auto cvt = [&](const float* src, bf16* dst, size_t n) {
    f32_to_bf16_kernel<<<cvtG, cvtB, 0, stream>>>(src, dst, n);
  };
  // 1) weights + V to bf16
  cvt(Wq, wq_b, (size_t)QKW * DHID);
  cvt(Wk, wk_b, (size_t)QKW * DHID);
  cvt(Wv, wv_b, (size_t)DHID * DHID);
  cvt(Wvv, wvv_b, (size_t)DHID * DHID);
  cvt(Wo, wo_b, (size_t)DHID * DHID);
  cvt(Wvo, wvo_b, (size_t)DHID * DHID);
  cvt(Wlv, wlv_b, (size_t)1024 * DHID);
  cvt(W1, w1_b, (size_t)DFF * 1024);
  cvt(W2, w2_b, (size_t)1024 * DFF);
  cvt(V, vin_b, (size_t)BN3 * DHID);

  // 2) ln1
  layernorm_kernel<<<dim3(BN), dim3(32), 0, stream>>>(H, ln1g, ln1b, hn_b, nullptr);

  // 3) projections (all bf16 WMMA)
  auto gemm = [&](const bf16* A, const bf16* W, const float* bias,
                  const float* resid, float* Cf, bf16* Cb, int M, int N, int K,
                  int act) {
    gemm_bf16_kernel<<<dim3(N / 64, M / 256), dim3(256), 0, stream>>>(
        A, W, bias, resid, Cf, Cb, M, N, K, act);
  };
  gemm(hn_b, wq_b, bq, nullptr, nullptr, q_b, BN, QKW, DHID, 0);
  gemm(hn_b, wk_b, bk, nullptr, nullptr, k_b, BN, QKW, DHID, 0);
  gemm(hn_b, wv_b, bv, nullptr, hv_f, nullptr, BN, DHID, DHID, 0);
  gemm(vin_b, wvv_b, nullptr, nullptr, vv_f, nullptr, BN3, DHID, DHID, 0);

  // 4) V_attn assembly, flash attention (double-buffered TDM), split
  assemble_vattn_kernel<<<dim3(4096), dim3(256), 0, stream>>>(hv_f, vv_f, vatt_b);
  attention_kernel<<<dim3(BB * NH * (NSEQ / 64)), dim3(128), 0, stream>>>(
      q_b, k_b, vatt_b, rbf, Dm, msk, res_f);
  split_res_kernel<<<dim3(4096), dim3(256), 0, stream>>>(res_f, hres_b, vres_b);

  // 5) output projections + residuals
  gemm(hres_b, wo_b, bo, H, h1_f, nullptr, BN, DHID, DHID, 0);
  gemm(vres_b, wvo_b, nullptr, V, vn_f, nullptr, BN3, DHID, DHID, 0);
  cvt(vn_f, vn_b, (size_t)BN3 * DHID);

  // 6) second half: ln2, Wlv, scaler, FFN, gated output
  layernorm_kernel<<<dim3(BN), dim3(32), 0, stream>>>(h1_f, ln2g, ln2b, nullptr,
                                                      hn2_f);
  gemm(vn_b, wlv_b, nullptr, nullptr, vpr_f, nullptr, BN3, 1024, DHID, 0);
  scaler_kernel<<<dim3(2048), dim3(256), 0, stream>>>(hn2_f, vpr_f, scl_b);
  gemm(scl_b, w1_b, b1, nullptr, nullptr, s1_b, BN, DFF, 1024, 1);  // SiLU
  gemm(s1_b, w2_b, b2, nullptr, so_f, nullptr, BN, 1024, DFF, 0);
  final_kernel<<<dim3(4096), dim3(256), 0, stream>>>(so_f, h1_f, vpr_f, vn_f, out);
}
